// OptimizedSparseAttentionLayer_91216515432606
// MI455X (gfx1250) — matrix operand order repaired and checked by reading, not compiled, not run on hardware
//
#include <hip/hip_runtime.h>

// ---------------------------------------------------------------------------
// Types / helpers
// ---------------------------------------------------------------------------
typedef __attribute__((ext_vector_type(16))) __bf16 bf16x16;
typedef __attribute__((ext_vector_type(8)))  __bf16 bf16x8;
typedef __attribute__((ext_vector_type(8)))  float  floatx8;

#define S_LEN 2048
#define D_DIM 1024
#define H_NUM 16
#define HD_DIM 64
#define DFF_DIM 4096
#define IH_NUM 4
#define ID_DIM 64
#define TOPK 64

__device__ __forceinline__ unsigned short f_to_bf(float f) {
    unsigned u = __builtin_bit_cast(unsigned, f);
    unsigned r = u + 0x7fffu + ((u >> 16) & 1u);
    return (unsigned short)(r >> 16);
}
__device__ __forceinline__ float bf_to_f(unsigned short u) {
    return __builtin_bit_cast(float, ((unsigned)u) << 16);
}
__device__ __forceinline__ __bf16 us_bf(unsigned short u) {
    return __builtin_bit_cast(__bf16, u);
}

// ---------------------------------------------------------------------------
// Tensor Data Mover (TDM) 2D tile load: global -> LDS, bf16 elements.
// D# group0/group1 packing per CDNA5 ISA 8.3/8.4.
// ---------------------------------------------------------------------------
#if __has_builtin(__builtin_amdgcn_tensor_load_to_lds)
#define HAVE_TDM 1
typedef unsigned int uint4v __attribute__((ext_vector_type(4)));
typedef int          int4v  __attribute__((ext_vector_type(4)));
typedef int          int8v  __attribute__((ext_vector_type(8)));

// tile0 = contiguous elements per row, tile1 = rows, stride0 = row pitch (elems)
// pad_int_code: pad after 2^(code+1) dwords; pad_amt_code: pad (code+1) dwords.
__device__ __forceinline__ void tdm_load_tile_bf16(
    unsigned lds_byte_off, const unsigned short* gptr,
    unsigned tile0, unsigned tile1, unsigned long long stride0,
    unsigned pad_int_code, unsigned pad_amt_code) {
    unsigned long long ga = (unsigned long long)(size_t)gptr;
    uint4v g0;
    g0[0] = 1u;                                           // count=1 (user D#)
    g0[1] = lds_byte_off;                                 // lds_addr
    g0[2] = (unsigned)ga;                                 // global_addr[31:0]
    g0[3] = (unsigned)((ga >> 32) & 0x01ffffffu) | (2u << 30);  // addr[56:32]|type=2
    unsigned pad_en = (pad_int_code | pad_amt_code) ? 1u : 0u;
    int8v g1;
    g1[0] = (int)((1u << 16) |                            // data_size=1 (2B)
                  (pad_en << 20) | (pad_int_code << 22) | (pad_amt_code << 25));
    g1[1] = (int)((tile0 & 0xffffu) << 16);               // tensor_dim0[15:0]
    g1[2] = (int)((tile0 >> 16) | ((tile1 & 0xffffu) << 16));  // dim0 hi | dim1 lo
    g1[3] = (int)((tile1 >> 16) | (tile0 << 16));         // dim1 hi | tile_dim0
    g1[4] = (int)(tile1 & 0xffffu);                       // tile_dim1 | tile_dim2=0
    g1[5] = (int)(unsigned)(stride0 & 0xffffffffu);       // tensor_dim0_stride lo
    g1[6] = (int)(unsigned)((stride0 >> 32) & 0xffffu);   // stride hi | dim1_stride=0
    g1[7] = 0;
    int4v gz = {0, 0, 0, 0};
#if __clang_major__ >= 23
    int8v gz8 = {0, 0, 0, 0, 0, 0, 0, 0};
    __builtin_amdgcn_tensor_load_to_lds(g0, g1, gz, gz, gz8, 0);
#else
    __builtin_amdgcn_tensor_load_to_lds(g0, g1, gz, gz, 0);
#endif
}
#else
#define HAVE_TDM 0
#endif

__device__ __forceinline__ unsigned lds_off_of(const void* p) {
    return (unsigned)(unsigned long long)(size_t)p;   // generic LDS addr: low 32 = offset
}

// ---------------------------------------------------------------------------
// fp32 [K,N] -> bf16 [K,Nw] cast with zero padding of columns N..Nw-1
// ---------------------------------------------------------------------------
__global__ void cast_pad_f32_bf16(const float* __restrict__ in,
                                  unsigned short* __restrict__ out,
                                  int K, int N, int Nw) {
    int i = blockIdx.x * 256 + threadIdx.x;
    if (i >= K * Nw) return;
    int row = i / Nw, col = i - row * Nw;
    out[i] = (col < N) ? f_to_bf(in[(size_t)row * N + col]) : (unsigned short)0;
}

// ---------------------------------------------------------------------------
// LayerNorm (row of 1024) -> bf16
// ---------------------------------------------------------------------------
__global__ __launch_bounds__(256) void layernorm_kernel(
    const float* __restrict__ x, const float* __restrict__ g,
    const float* __restrict__ b, unsigned short* __restrict__ out) {
    int row = blockIdx.x, tid = threadIdx.x;
    __shared__ float s1[256], s2[256];
    const float* xr = x + (size_t)row * D_DIM;
    float s = 0.f, q = 0.f;
    for (int i = tid; i < D_DIM; i += 256) { float v = xr[i]; s += v; q += v * v; }
    s1[tid] = s; s2[tid] = q;
    __syncthreads();
    for (int off = 128; off > 0; off >>= 1) {
        if (tid < off) { s1[tid] += s1[tid + off]; s2[tid] += s2[tid + off]; }
        __syncthreads();
    }
    float mean = s1[0] * (1.0f / D_DIM);
    float var  = s2[0] * (1.0f / D_DIM) - mean * mean;
    float r = rsqrtf(var + 1e-5f);
    for (int i = tid; i < D_DIM; i += 256) {
        float v = (xr[i] - mean) * r * g[i] + b[i];
        out[(size_t)row * D_DIM + i] = f_to_bf(v);
    }
}

// ---------------------------------------------------------------------------
// Generic bf16 WMMA GEMM: out = A[M,K] @ B[K,ldb(>=N)] (+bias)(gelu?)(+resid)
// Tile 128x128x32, 256 threads = 8 waves, wave computes 32x64.
// TDM=1: B tile staged by Tensor Data Mover (wave 0 issues, TENSORcnt wait).
// ---------------------------------------------------------------------------
#define BM 128
#define BN 128
#define BK 32
#define ASTR 40
#define BSTR 136   // 128 + 8 elems = 64 dwords + 4 pad dwords per row (TDM pad codes 5/3)

template <int TDM>
__global__ __launch_bounds__(256) void gemm_bf16_wmma(
    const unsigned short* __restrict__ A, const unsigned short* __restrict__ Bw,
    const float* __restrict__ bias, const float* __restrict__ resid,
    float* __restrict__ outF, unsigned short* __restrict__ outB,
    int M, int N, int K, int ldb, int gelu) {
    __shared__ __align__(16) unsigned short As[BM * ASTR];
    __shared__ __align__(16) unsigned short Bs[BK * BSTR];

    int tid = threadIdx.x;
    int lane = tid & 31, wid = tid >> 5;
    int wm = (wid & 3) * 32, wn = (wid >> 2) * 64;
    int tile_m = blockIdx.y * BM, tile_n = blockIdx.x * BN;

    // per-thread staging coordinates
    int ar = tid >> 2, ac = (tid & 3) << 3;          // A: row, col8 (+64 rows for 2nd)
    int bk = tid >> 4, bn = (tid & 15) << 3;         // B: krow, col8 (+16 rows for 2nd)

    floatx8 acc[2][4];
#pragma unroll
    for (int i = 0; i < 2; ++i)
#pragma unroll
        for (int j = 0; j < 4; ++j) acc[i][j] = {};

    for (int k0 = 0; k0 < K; k0 += BK) {
#if HAVE_TDM
        if (TDM && wid == 0) {
            // async DMA: 32 rows x 128 bf16 of B into Bs with 4-dword row padding
            tdm_load_tile_bf16(lds_off_of(Bs),
                               Bw + (size_t)k0 * ldb + tile_n,
                               BN, BK, (unsigned long long)ldb,
                               /*pad 64dw*/ 5, /*amt 4dw*/ 3);
        }
#endif
        // A tile [128 x 32]: batch 2 loads, then 2 LDS stores
        bf16x8 av0 = *reinterpret_cast<const bf16x8*>(A + (size_t)(tile_m + ar) * K + k0 + ac);
        bf16x8 av1 = *reinterpret_cast<const bf16x8*>(A + (size_t)(tile_m + ar + 64) * K + k0 + ac);
        if (!TDM || !HAVE_TDM) {
            bf16x8 bv0 = *reinterpret_cast<const bf16x8*>(Bw + (size_t)(k0 + bk) * ldb + tile_n + bn);
            bf16x8 bv1 = *reinterpret_cast<const bf16x8*>(Bw + (size_t)(k0 + bk + 16) * ldb + tile_n + bn);
            *reinterpret_cast<bf16x8*>(&Bs[bk * BSTR + bn]) = bv0;
            *reinterpret_cast<bf16x8*>(&Bs[(bk + 16) * BSTR + bn]) = bv1;
        }
        *reinterpret_cast<bf16x8*>(&As[ar * ASTR + ac]) = av0;
        *reinterpret_cast<bf16x8*>(&As[(ar + 64) * ASTR + ac]) = av1;

        // prefetch next K tile toward WGP while WMMAs run
        if (k0 + BK < K) {
            __builtin_prefetch(A + (size_t)(tile_m + ar) * K + k0 + BK + ac, 0, 1);
            if (!TDM || !HAVE_TDM)
                __builtin_prefetch(Bw + (size_t)(k0 + BK + bk) * ldb + tile_n + bn, 0, 1);
        }
#if HAVE_TDM
        if (TDM && wid == 0) __builtin_amdgcn_s_wait_tensorcnt(0);
#endif
        __syncthreads();

        // A fragments (16-bit A layout: lanes 0-15 K{0..7,16..23}, 16-31 K{8..15,24..31})
        bf16x16 afrag[2], bfrag[4];
        int mrow = lane & 15;
        int kg = (lane < 16) ? 0 : 8;
#pragma unroll
        for (int i = 0; i < 2; ++i) {
            int m = wm + i * 16 + mrow;
            bf16x8 lo = *reinterpret_cast<const bf16x8*>(&As[m * ASTR + kg]);
            bf16x8 hi = *reinterpret_cast<const bf16x8*>(&As[m * ASTR + kg + 16]);
#pragma unroll
            for (int e = 0; e < 8; ++e) { afrag[i][e] = lo[e]; afrag[i][e + 8] = hi[e]; }
        }
        // B fragments (lane%16 = N column, K{0..7,16..23} / K{8..15,24..31} per lane half)
#pragma unroll
        for (int j = 0; j < 4; ++j) {
            int n0 = wn + j * 16;
#pragma unroll
            for (int e = 0; e < 8; ++e) {
                bfrag[j][e] = us_bf(Bs[(kg + e) * BSTR + n0 + mrow]);
                bfrag[j][e + 8] = us_bf(Bs[(kg + 16 + e) * BSTR + n0 + mrow]);
            }
        }
#pragma unroll
        for (int i = 0; i < 2; ++i)
#pragma unroll
            for (int j = 0; j < 4; ++j)
                acc[i][j] = __builtin_amdgcn_wmma_f32_16x16x32_bf16(
                    false, afrag[i], false, bfrag[j], (short)0, acc[i][j], false, false);
        __syncthreads();
    }

    // epilogue: C VGPR r -> (M = r or r+8, N = lane%16)
    int mhalf = (lane < 16) ? 0 : 8;
    int ncol0 = tile_n + wn + (lane & 15);
#pragma unroll
    for (int i = 0; i < 2; ++i) {
#pragma unroll
        for (int j = 0; j < 4; ++j) {
#pragma unroll
            for (int r = 0; r < 8; ++r) {
                int row = tile_m + wm + i * 16 + mhalf + r;
                int col = ncol0 + j * 16;
                if (col < N) {
                    float v = acc[i][j][r];
                    if (bias)  v += bias[col];
                    if (gelu)  v = 0.5f * v * (1.0f + erff(v * 0.70710678f));
                    if (resid) v += resid[(size_t)row * N + col];
                    if (outF)  outF[(size_t)row * N + col] = v;
                    if (outB)  outB[(size_t)row * N + col] = f_to_bf(v);
                }
            }
        }
    }
}

// ---------------------------------------------------------------------------
// RoPE on q,k (interleaved pairs) + split qkv -> bf16 q/k/v
// ---------------------------------------------------------------------------
__global__ void rope_split_kernel(const float* __restrict__ qkv,
                                  unsigned short* __restrict__ qb,
                                  unsigned short* __restrict__ kb,
                                  unsigned short* __restrict__ vb) {
    int id = blockIdx.x * 256 + threadIdx.x;          // over S * D/2
    if (id >= S_LEN * (D_DIM / 2)) return;
    int s = id >> 9, r = id & 511;
    int h = r >> 5, p = r & 31;
    const float* base = qkv + (size_t)s * (3 * D_DIM);
    int o = h * HD_DIM + 2 * p;
    float q0 = base[o], q1 = base[o + 1];
    float k0 = base[D_DIM + o], k1 = base[D_DIM + o + 1];
    float theta = __powf(10000.0f, -(float)(2 * p) / (float)HD_DIM);
    float ang = (float)s * theta;
    float sn, cs;
    __sincosf(ang, &sn, &cs);
    size_t ro = (size_t)s * D_DIM + o;
    qb[ro]     = f_to_bf(q0 * cs - q1 * sn);
    qb[ro + 1] = f_to_bf(q1 * cs + q0 * sn);
    kb[ro]     = f_to_bf(k0 * cs - k1 * sn);
    kb[ro + 1] = f_to_bf(k1 * cs + k0 * sn);
    vb[ro]     = f_to_bf(base[2 * D_DIM + o]);
    vb[ro + 1] = f_to_bf(base[2 * D_DIM + o + 1]);
}

// ---------------------------------------------------------------------------
// Indexer scores: I[t,s] = sum_h wi[t,h] * relu(qi[t,h,:] . ki[s,:])
// One block = 16x16 (t,s) tile, 4 waves (one per indexer head), WMMA over ID.
// ---------------------------------------------------------------------------
#define KI_STR 128
#define WI_STR 128

__global__ __launch_bounds__(128) void idx_score_kernel(
    const unsigned short* __restrict__ qi, const unsigned short* __restrict__ ki,
    const float* __restrict__ wi, float* __restrict__ scores) {
    int sbase = blockIdx.x * 16, tbase = blockIdx.y * 16;
    if (sbase > tbase + 15) return;                   // fully above diagonal

    __shared__ __align__(16) unsigned short kis[16 * ID_DIM];
    __shared__ float accs[256];
    int tid = threadIdx.x, lane = tid & 31, h = tid >> 5;

    {   // stage ki tile [16 x 64]
        int r = tid >> 3, c = (tid & 7) << 3;
        *reinterpret_cast<bf16x8*>(&kis[r * ID_DIM + c]) =
            *reinterpret_cast<const bf16x8*>(ki + (size_t)(sbase + r) * KI_STR + c);
    }
    accs[tid] = 0.f;
    accs[tid + 128] = 0.f;
    __syncthreads();

    floatx8 c = {};
    int mrow = lane & 15;
#pragma unroll
    for (int ks = 0; ks < 2; ++ks) {
        int kg = ((lane < 16) ? 0 : 8) + ks * 32;
        int t = tbase + mrow;
        const unsigned short* qrow = qi + (size_t)t * (IH_NUM * ID_DIM) + h * ID_DIM;
        bf16x8 lo = *reinterpret_cast<const bf16x8*>(qrow + kg);
        bf16x8 hi = *reinterpret_cast<const bf16x8*>(qrow + kg + 16);
        bf16x16 a, b;
#pragma unroll
        for (int e = 0; e < 8; ++e) { a[e] = lo[e]; a[e + 8] = hi[e]; }
#pragma unroll
        for (int e = 0; e < 16; ++e) b[e] = us_bf(kis[mrow * ID_DIM + kg + (e & 7) + ((e >> 3) << 4)]);
        c = __builtin_amdgcn_wmma_f32_16x16x32_bf16(false, a, false, b, (short)0, c,
                                                    false, false);
    }
#pragma unroll
    for (int r = 0; r < 8; ++r) {
        int m = ((lane < 16) ? 0 : 8) + r;
        int t = tbase + m;
        float v = c[r];
        v = v > 0.f ? v : 0.f;
        v *= wi[(size_t)t * WI_STR + h];
        atomicAdd(&accs[m * 16 + (lane & 15)], v);
    }
    __syncthreads();
    for (int idx = tid; idx < 256; idx += 128) {
        int m = idx >> 4, n = idx & 15;
        int t = tbase + m, s = sbase + n;
        scores[(size_t)t * S_LEN + s] = (s <= t) ? accs[idx] : -3.0e38f;
    }
}

// ---------------------------------------------------------------------------
// Per-row top-64 (causal): iterative argmax in LDS
// ---------------------------------------------------------------------------
__global__ __launch_bounds__(256) void topk_kernel(const float* __restrict__ scores,
                                                   int* __restrict__ topk) {
    int t = blockIdx.x, tid = threadIdx.x;
    __shared__ float row[S_LEN];
    __shared__ float smax[256];
    __shared__ int   simx[256];
    for (int s = tid; s < S_LEN; s += 256)
        row[s] = (s <= t) ? scores[(size_t)t * S_LEN + s] : -3.4e38f;
    __syncthreads();
    for (int k = 0; k < TOPK; ++k) {
        float bv = -3.4e38f; int bi = 0;
        for (int s = tid; s < S_LEN; s += 256) {
            float v = row[s];
            if (v > bv) { bv = v; bi = s; }
        }
        smax[tid] = bv; simx[tid] = bi;
        __syncthreads();
        for (int off = 128; off > 0; off >>= 1) {
            if (tid < off && smax[tid + off] > smax[tid]) {
                smax[tid] = smax[tid + off]; simx[tid] = simx[tid + off];
            }
            __syncthreads();
        }
        if (tid == 0) { topk[(size_t)t * TOPK + k] = simx[0]; row[simx[0]] = -3.4e38f; }
        __syncthreads();
    }
}

// ---------------------------------------------------------------------------
// Sparse SDPA over top-64 gathered keys. One wave per (t, h).
// ---------------------------------------------------------------------------
__global__ __launch_bounds__(256) void sattn_kernel(
    const unsigned short* __restrict__ qb, const unsigned short* __restrict__ kb,
    const unsigned short* __restrict__ vb, const int* __restrict__ topk,
    unsigned short* __restrict__ ao) {
    int tid = threadIdx.x, lane = tid & 31, wid = tid >> 5;
    int wg = blockIdx.x * 8 + wid;
    int t = wg >> 4, h = wg & 15;

    __shared__ float qsh[8][HD_DIM];
    __shared__ float psh[8][TOPK];
    __shared__ int   ish[8][TOPK];

    const unsigned short* qrow = qb + (size_t)t * D_DIM + h * HD_DIM;
    qsh[wid][lane]      = bf_to_f(qrow[lane]);
    qsh[wid][lane + 32] = bf_to_f(qrow[lane + 32]);
    ish[wid][lane]      = topk[(size_t)t * TOPK + lane];
    ish[wid][lane + 32] = topk[(size_t)t * TOPK + lane + 32];
    __syncthreads();

    float sc[2];
#pragma unroll
    for (int cc = 0; cc < 2; ++cc) {
        int j = lane + cc * 32;
        int s = ish[wid][j];
        const unsigned short* krow = kb + (size_t)s * D_DIM + h * HD_DIM;
        float acc = 0.f;
        for (int d = 0; d < HD_DIM; ++d) acc += qsh[wid][d] * bf_to_f(krow[d]);
        sc[cc] = (s <= t) ? acc * 0.125f : -3.0e38f;
    }
    float m = fmaxf(sc[0], sc[1]);
    for (int off = 16; off > 0; off >>= 1) m = fmaxf(m, __shfl_xor(m, off, 32));
    float e0 = __expf(sc[0] - m), e1 = __expf(sc[1] - m);
    float sum = e0 + e1;
    for (int off = 16; off > 0; off >>= 1) sum += __shfl_xor(sum, off, 32);
    float inv = 1.0f / sum;
    psh[wid][lane] = e0 * inv; psh[wid][lane + 32] = e1 * inv;
    __syncthreads();

    float a0 = 0.f, a1 = 0.f;
    for (int j = 0; j < TOPK; ++j) {
        float p = psh[wid][j];
        int s = ish[wid][j];
        const unsigned short* vrow = vb + (size_t)s * D_DIM + h * HD_DIM;
        a0 += p * bf_to_f(vrow[lane]);
        a1 += p * bf_to_f(vrow[lane + 32]);
    }
    unsigned short* orow = ao + (size_t)t * D_DIM + h * HD_DIM;
    orow[lane]      = f_to_bf(a0);
    orow[lane + 32] = f_to_bf(a1);
}

// ---------------------------------------------------------------------------
// Host launcher
// ---------------------------------------------------------------------------
extern "C" void kernel_launch(void* const* d_in, const int* in_sizes, int n_in,
                              void* d_out, int out_size, void* d_ws, size_t ws_size,
                              hipStream_t stream) {
    (void)in_sizes; (void)n_in; (void)out_size; (void)ws_size;
    const float* x      = (const float*)d_in[0];
    const float* w_qkv  = (const float*)d_in[1];
    const float* w_o    = (const float*)d_in[2];
    const float* ffn_w1 = (const float*)d_in[3];
    const float* ffn_b1 = (const float*)d_in[4];
    const float* ffn_w2 = (const float*)d_in[5];
    const float* ffn_b2 = (const float*)d_in[6];
    const float* ln1_g  = (const float*)d_in[7];
    const float* ln1_b  = (const float*)d_in[8];
    const float* ln2_g  = (const float*)d_in[9];
    const float* ln2_b  = (const float*)d_in[10];
    const float* idx_wq = (const float*)d_in[11];
    const float* idx_wk = (const float*)d_in[12];
    const float* idx_ww = (const float*)d_in[13];

    size_t off = 0;
    auto alloc = [&](size_t bytes) -> void* {
        off = (off + 255) & ~(size_t)255;
        void* p = (char*)d_ws + off;
        off += bytes;
        return p;
    };
    unsigned short* nxB   = (unsigned short*)alloc((size_t)S_LEN * D_DIM * 2);
    unsigned short* wqkvB = (unsigned short*)alloc((size_t)D_DIM * 3 * D_DIM * 2);
    unsigned short* woB   = (unsigned short*)alloc((size_t)D_DIM * D_DIM * 2);
    unsigned short* w1B   = (unsigned short*)alloc((size_t)D_DIM * DFF_DIM * 2);
    unsigned short* w2B   = (unsigned short*)alloc((size_t)DFF_DIM * D_DIM * 2);
    unsigned short* wiqB  = (unsigned short*)alloc((size_t)D_DIM * 256 * 2);
    unsigned short* wikB  = (unsigned short*)alloc((size_t)D_DIM * 128 * 2);  // padded 64->128
    unsigned short* wiwB  = (unsigned short*)alloc((size_t)D_DIM * 128 * 2);  // padded 4->128
    float*          qkvF  = (float*)alloc((size_t)S_LEN * 3 * D_DIM * 4);
    unsigned short* qB    = (unsigned short*)alloc((size_t)S_LEN * D_DIM * 2);
    unsigned short* kB    = (unsigned short*)alloc((size_t)S_LEN * D_DIM * 2);
    unsigned short* vB    = (unsigned short*)alloc((size_t)S_LEN * D_DIM * 2);
    unsigned short* qiB   = (unsigned short*)alloc((size_t)S_LEN * 256 * 2);
    unsigned short* kiB   = (unsigned short*)alloc((size_t)S_LEN * KI_STR * 2);
    float*          wiF   = (float*)alloc((size_t)S_LEN * WI_STR * 4);
    float*          scoF  = (float*)alloc((size_t)S_LEN * S_LEN * 4);
    int*            tkI   = (int*)alloc((size_t)S_LEN * TOPK * 4);
    unsigned short* aoB   = (unsigned short*)alloc((size_t)S_LEN * D_DIM * 2);
    float*          x1F   = (float*)alloc((size_t)S_LEN * D_DIM * 4);
    unsigned short* hB    = (unsigned short*)alloc((size_t)S_LEN * D_DIM * 2);
    unsigned short* f1B   = (unsigned short*)alloc((size_t)S_LEN * DFF_DIM * 2);

    auto cg = [](int n) { return (n + 255) / 256; };
    // weight casts (padded to column stride multiple of 128)
    cast_pad_f32_bf16<<<cg(D_DIM * 3 * D_DIM), 256, 0, stream>>>(w_qkv, wqkvB, D_DIM, 3 * D_DIM, 3 * D_DIM);
    cast_pad_f32_bf16<<<cg(D_DIM * D_DIM), 256, 0, stream>>>(w_o, woB, D_DIM, D_DIM, D_DIM);
    cast_pad_f32_bf16<<<cg(D_DIM * DFF_DIM), 256, 0, stream>>>(ffn_w1, w1B, D_DIM, DFF_DIM, DFF_DIM);
    cast_pad_f32_bf16<<<cg(DFF_DIM * D_DIM), 256, 0, stream>>>(ffn_w2, w2B, DFF_DIM, D_DIM, D_DIM);
    cast_pad_f32_bf16<<<cg(D_DIM * 256), 256, 0, stream>>>(idx_wq, wiqB, D_DIM, 256, 256);
    cast_pad_f32_bf16<<<cg(D_DIM * 128), 256, 0, stream>>>(idx_wk, wikB, D_DIM, ID_DIM, 128);
    cast_pad_f32_bf16<<<cg(D_DIM * 128), 256, 0, stream>>>(idx_ww, wiwB, D_DIM, IH_NUM, 128);

    // LN1
    layernorm_kernel<<<S_LEN, 256, 0, stream>>>(x, ln1_g, ln1_b, nxB);

    // QKV projection (TDM-staged weights)
    gemm_bf16_wmma<1><<<dim3(3 * D_DIM / BN, S_LEN / BM), 256, 0, stream>>>(
        nxB, wqkvB, nullptr, nullptr, qkvF, nullptr, S_LEN, 3 * D_DIM, D_DIM, 3 * D_DIM, 0);
    // RoPE + split
    rope_split_kernel<<<S_LEN * (D_DIM / 2) / 256, 256, 0, stream>>>(qkvF, qB, kB, vB);

    // indexer projections (vector-staged path)
    gemm_bf16_wmma<0><<<dim3(256 / BN, S_LEN / BM), 256, 0, stream>>>(
        nxB, wiqB, nullptr, nullptr, nullptr, qiB, S_LEN, 256, D_DIM, 256, 0);
    gemm_bf16_wmma<0><<<dim3(1, S_LEN / BM), 256, 0, stream>>>(
        nxB, wikB, nullptr, nullptr, nullptr, kiB, S_LEN, KI_STR, D_DIM, 128, 0);
    gemm_bf16_wmma<0><<<dim3(1, S_LEN / BM), 256, 0, stream>>>(
        nxB, wiwB, nullptr, nullptr, wiF, nullptr, S_LEN, WI_STR, D_DIM, 128, 0);

    // indexer scores, top-k, sparse attention
    idx_score_kernel<<<dim3(S_LEN / 16, S_LEN / 16), 128, 0, stream>>>(qiB, kiB, wiF, scoF);
    topk_kernel<<<S_LEN, 256, 0, stream>>>(scoF, tkI);
    sattn_kernel<<<S_LEN * H_NUM / 8, 256, 0, stream>>>(qB, kB, vB, tkI, aoB);

    // output projection + residual (TDM)
    gemm_bf16_wmma<1><<<dim3(D_DIM / BN, S_LEN / BM), 256, 0, stream>>>(
        aoB, woB, nullptr, x, x1F, nullptr, S_LEN, D_DIM, D_DIM, D_DIM, 0);

    // LN2 + FFN (TDM)
    layernorm_kernel<<<S_LEN, 256, 0, stream>>>(x1F, ln2_g, ln2_b, hB);
    gemm_bf16_wmma<1><<<dim3(DFF_DIM / BN, S_LEN / BM), 256, 0, stream>>>(
        hB, w1B, ffn_b1, nullptr, nullptr, f1B, S_LEN, DFF_DIM, D_DIM, DFF_DIM, 1);
    gemm_bf16_wmma<1><<<dim3(D_DIM / BN, S_LEN / BM), 256, 0, stream>>>(
        f1B, w2B, ffn_b2, x1F, (float*)d_out, nullptr, S_LEN, D_DIM, DFF_DIM, D_DIM, 0);
}
